// DGCNN_Grouper_Res_1967095021878
// MI455X (gfx1250) — matrix operand order repaired and checked by reading, not compiled, not run on hardware
//
#include <hip/hip_runtime.h>
#include <hip/hip_bf16.h>

// ---------------------------------------------------------------------------
// DGCNN grouper for MI455X (gfx1250, wave32, WMMA bf16 16x16x32).
// bf16 activations/weights, f32 WMMA accumulation, fused max/stat epilogues.
// A-tiles staged into LDS via GLOBAL_LOAD_ASYNC_TO_LDS_B128 (ASYNCcnt path);
// each wave blocks 4 row-tiles to reuse every B fragment 4x.
// ---------------------------------------------------------------------------

typedef __attribute__((ext_vector_type(16))) __bf16 v16bf;
typedef __attribute__((ext_vector_type(8)))  float  v8f;
typedef __attribute__((ext_vector_type(4)))  unsigned int v4u;
typedef unsigned short u16t;
typedef unsigned int   u32t;

#define BATCH 8
#define NPTS  2048
#define KNN   16
// edge rows: BATCH*NPTS*KNN = 262144 ; point rows: BATCH*NPTS = 16384

static __device__ __forceinline__ u16t f32_to_bf16(float f) {
    u32t u = __float_as_uint(f);
    u += 0x7FFFu + ((u >> 16) & 1u);          // round-to-nearest-even
    return (u16t)(u >> 16);
}
static __device__ __forceinline__ float bf16_to_f32(u16t h) {
    return __uint_as_float(((u32t)h) << 16);
}

union Frag16 { v16bf v; v4u q[2]; };

// ---------------------------------------------------------------------------
// init: zero groupnorm stat partials, gmax accumulator, global_feat output
// ---------------------------------------------------------------------------
__global__ void init_kernel(float* stats, int* gmax, float* outGF) {
    int i = blockIdx.x * 256 + threadIdx.x;          // grid 16 x 256 -> 4096
    if (i < BATCH * 4 * 32 * 2) stats[i] = 0.f;
    if (i < BATCH * 512) { gmax[i] = 0; outGF[i] = 0.f; }
}

// ---------------------------------------------------------------------------
// f32 -> bf16 weight conversion
// ---------------------------------------------------------------------------
__global__ void cvt_kernel(const float* __restrict__ s, u16t* __restrict__ d, int n) {
    int i = blockIdx.x * 256 + threadIdx.x;
    if (i < n) d[i] = f32_to_bf16(s[i]);
}

// ---------------------------------------------------------------------------
// h = W_it @ f + b_it   (3 -> 64), f32.  grid B*N, block 64.
// ---------------------------------------------------------------------------
__global__ void h_kernel(const float* __restrict__ f, const float* __restrict__ W_it,
                         const float* __restrict__ b_it, float* __restrict__ h) {
    int bn = blockIdx.x;
    int b  = bn >> 11;                  // / NPTS
    int n  = bn & (NPTS - 1);
    int o  = threadIdx.x;
    const float* xb = f + (size_t)b * 3 * NPTS;
    float x0 = xb[n], x1 = xb[NPTS + n], x2 = xb[2 * NPTS + n];
    float v = b_it[o] + x0 * W_it[o * 3 + 0] + x1 * W_it[o * 3 + 1] + x2 * W_it[o * 3 + 2];
    h[(size_t)bn * 64 + o] = v;
}

// ---------------------------------------------------------------------------
// brute-force KNN (K=16 smallest distances incl. self). grid 64, block 256.
// blockIdx.x: b = >>3, chunk of 256 points = &7. Coords cached in LDS.
// ---------------------------------------------------------------------------
__global__ __launch_bounds__(256)
void knn_kernel(const float* __restrict__ x, int* __restrict__ idx) {
    __shared__ float xs[NPTS], ys[NPTS], zs[NPTS], sq[NPTS];
    int b = blockIdx.x >> 3;
    const float* xb = x + (size_t)b * 3 * NPTS;
    for (int i = threadIdx.x; i < NPTS; i += 256) {
        float a = xb[i], bb2 = xb[NPTS + i], c = xb[2 * NPTS + i];
        xs[i] = a; ys[i] = bb2; zs[i] = c;
        sq[i] = a * a + bb2 * bb2 + c * c;
    }
    __syncthreads();

    int p = (blockIdx.x & 7) * 256 + threadIdx.x;
    float px = xs[p], py = ys[p], pz = zs[p], ps = sq[p];

    float bd[KNN]; int bi[KNN];
    #pragma unroll
    for (int k = 0; k < KNN; ++k) { bd[k] = 3.4e38f; bi[k] = 0; }
    float worst = 3.4e38f;

    for (int j = 0; j < NPTS; ++j) {
        float d = ps + sq[j] - 2.f * (px * xs[j] + py * ys[j] + pz * zs[j]);
        if (d < worst) {
            int pos = 0; float w = bd[0];
            #pragma unroll
            for (int k = 1; k < KNN; ++k) if (bd[k] > w) { w = bd[k]; pos = k; }
            #pragma unroll
            for (int k = 0; k < KNN; ++k) if (k == pos) { bd[k] = d; bi[k] = j; }
            w = bd[0];
            #pragma unroll
            for (int k = 1; k < KNN; ++k) w = fmaxf(w, bd[k]);
            worst = w;
        }
    }
    int* out = idx + (size_t)((b << 11) + p) * KNN;
    #pragma unroll
    for (int k = 0; k < KNN; ++k) out[k] = bi[k];
}

// ---------------------------------------------------------------------------
// edge features: feat[row] = [h[nb]-h[n] (64) | h[n] (64)]  bf16.
// grid (B*N*K)/4, block 256 (4 rows x 64 channels).
// ---------------------------------------------------------------------------
__global__ void feat_kernel(const float* __restrict__ h, const int* __restrict__ idx,
                            u16t* __restrict__ feat) {
    int row = blockIdx.x * 4 + (threadIdx.x >> 6);
    int c   = threadIdx.x & 63;
    int bn  = row >> 4;            // (b*N + n)
    int b   = row >> 15;           // / (N*K)
    int nb  = idx[row];
    float hc = h[(size_t)bn * 64 + c];
    float hn = h[((size_t)(b << 11) + nb) * 64 + c];
    feat[(size_t)row * 128 + c]      = f32_to_bf16(hn - hc);
    feat[(size_t)row * 128 + 64 + c] = f32_to_bf16(hc);
}

// ---------------------------------------------------------------------------
// Generic WMMA bf16 GEMM:  out[r][o] = act( sum_c A[r][c]*W[o][c] + bias[o] )
//   grid (rows/64, Ocols/64), block 128 (4 waves).
//   Block stages a 64 x Ck A-tile into LDS via async direct-to-LDS b128 ops;
//   wave w owns one 16-col tile and iterates 4 row tiles (B-frag reuse 4x).
// Epilogue flags: bf16 store, f32 transposed store, per-batch column max
// (atomic, relu outputs >= 0 so int-bit compare is valid), groupnorm stats.
// ---------------------------------------------------------------------------
__global__ __launch_bounds__(128)
void gemm_bf16(const u16t* __restrict__ A, const u16t* __restrict__ W,
               const float* __restrict__ bias, int Ck, int Ocols,
               u16t* __restrict__ outB,
               float* __restrict__ outT, int Nper,
               int* __restrict__ rowmax, int rowsPerBatch,
               float* __restrict__ statsP, int relu)
{
    extern __shared__ u16t aT[];                // 64 x (Ck + 8) bf16
    const int strideA = Ck + 8;
    const int r0 = blockIdx.x * 64;

    {   // async stage: one GLOBAL_LOAD_ASYNC_TO_LDS_B128 per 16B chunk per lane
        const int chunksPerRow = Ck >> 3;       // 8 bf16 per chunk
        const int totalChunks  = 64 * chunksPerRow;   // multiple of 128 for all Ck
        const unsigned long long abase =
            (unsigned long long)(uintptr_t)(A + (size_t)r0 * Ck);
        for (int ch = threadIdx.x; ch < totalChunks; ch += 128) {
            int rr = ch / chunksPerRow;
            int cc = (ch - rr * chunksPerRow) << 3;
            unsigned ldsOff = (unsigned)(uintptr_t)&aT[rr * strideA + cc];
            unsigned gOff   = (unsigned)((rr * Ck + cc) * 2);
            asm volatile("global_load_async_to_lds_b128 %0, %1, %2"
                         :: "v"(ldsOff), "v"(gOff), "s"(abase)
                         : "memory");
        }
        asm volatile("s_wait_asynccnt 0x0" ::: "memory");
    }
    __syncthreads();

    const int lane = threadIdx.x & 31;
    const int wv   = threadIdx.x >> 5;
    const int ct   = blockIdx.y * 4 + wv;       // 16-wide column tile
    const int col  = ct * 16 + (lane & 15);
    const int hi   = lane >> 4;

    // A fragment (ISA interleaved layout): lanes0-15 K{0..7,16..23}, lanes16-31 K{8..15,24..31}
    const u16t* ab0 = &aT[(lane & 15) * strideA + hi * 8];
    // B fragment: lane column = col, same interleaved layout: lanes0-15 K{0..7,16..23}, lanes16-31 K{8..15,24..31}
    const u16t* wb = W + (size_t)col * Ck + hi * 8;

    v8f acc[4];
    #pragma unroll
    for (int rt = 0; rt < 4; ++rt) acc[rt] = (v8f){0.f,0.f,0.f,0.f,0.f,0.f,0.f,0.f};

    for (int kk = 0; kk < Ck; kk += 32) {
        Frag16 b;
        b.q[0] = *(const v4u*)(wb + kk);
        b.q[1] = *(const v4u*)(wb + kk + 16);
        #pragma unroll
        for (int rt = 0; rt < 4; ++rt) {
            Frag16 a;
            const u16t* ap = ab0 + rt * 16 * strideA + kk;
            a.q[0] = *(const v4u*)(ap);
            a.q[1] = *(const v4u*)(ap + 16);
            acc[rt] = __builtin_amdgcn_wmma_f32_16x16x32_bf16(false, a.v, false, b.v,
                                                              (short)0, acc[rt], false, false);
        }
    }

    if (bias) {
        float bo = bias[col];
        #pragma unroll
        for (int rt = 0; rt < 4; ++rt)
            #pragma unroll
            for (int i = 0; i < 8; ++i) acc[rt][i] += bo;
    }
    if (relu) {
        #pragma unroll
        for (int rt = 0; rt < 4; ++rt)
            #pragma unroll
            for (int i = 0; i < 8; ++i) acc[rt][i] = fmaxf(acc[rt][i], 0.f);
    }
    if (statsP) {   // groupnorm partials; all 4 row tiles share batch & group
        float s = 0.f, s2 = 0.f;
        #pragma unroll
        for (int rt = 0; rt < 4; ++rt)
            #pragma unroll
            for (int i = 0; i < 8; ++i) { s += acc[rt][i]; s2 += acc[rt][i] * acc[rt][i]; }
        #pragma unroll
        for (int off = 16; off >= 1; off >>= 1) {
            s  += __shfl_xor(s,  off, 32);
            s2 += __shfl_xor(s2, off, 32);
        }
        if (lane == 0) {
            int bb = r0 / rowsPerBatch;
            int g  = (ct * 16) >> 5;
            float* sp = statsP + (size_t)(((bb * 4 + g) * 32 + (blockIdx.x & 31)) * 2);
            atomicAdd(sp, s);
            atomicAdd(sp + 1, s2);
        }
    }
    if (outB) {     // bf16 row-major store (D layout: row = i + 8*hi, col = lane%16)
        #pragma unroll
        for (int rt = 0; rt < 4; ++rt)
            #pragma unroll
            for (int i = 0; i < 8; ++i) {
                int m = rt * 16 + i + hi * 8;
                outB[(size_t)(r0 + m) * Ocols + col] = f32_to_bf16(acc[rt][i]);
            }
    }
    if (outT) {     // f32 transposed store: out[b][col][n]
        #pragma unroll
        for (int rt = 0; rt < 4; ++rt)
            #pragma unroll
            for (int i = 0; i < 8; ++i) {
                int gr = r0 + rt * 16 + i + hi * 8;
                int bb = gr / Nper;
                int nn = gr - bb * Nper;
                outT[((size_t)bb * Ocols + col) * Nper + nn] = acc[rt][i];
            }
    }
    if (rowmax) {   // fused max over rows per batch (64 rows share a batch)
        float m = acc[0][0];
        #pragma unroll
        for (int rt = 0; rt < 4; ++rt)
            #pragma unroll
            for (int i = 0; i < 8; ++i) m = fmaxf(m, acc[rt][i]);
        m = fmaxf(m, __shfl_xor(m, 16, 32));
        if (lane < 16) {
            int bb = r0 / rowsPerBatch;
            atomicMax(&rowmax[(size_t)bb * Ocols + col], __float_as_int(m));
        }
    }
}

// ---------------------------------------------------------------------------
// groupnorm + leakyrelu(0.2) applied in place to y (bf16), and K-max into
// cat[:, 0:128] (local_base).  grid B*N, block 128 (one channel per thread).
// ---------------------------------------------------------------------------
__global__ void norm_kernel(u16t* __restrict__ y, const float* __restrict__ statsP,
                            const float* __restrict__ gn_g, const float* __restrict__ gn_b,
                            u16t* __restrict__ cat) {
    int bn = blockIdx.x;
    int b  = bn >> 11;
    int c  = threadIdx.x;
    int g  = c >> 5;
    float s = 0.f, s2 = 0.f;
    const float* sp = statsP + (size_t)(b * 4 + g) * 32 * 2;
    for (int p = 0; p < 32; ++p) { s += sp[p * 2]; s2 += sp[p * 2 + 1]; }
    const float cnt = (float)NPTS * KNN * 32;
    float mu  = s / cnt;
    float var = s2 / cnt - mu * mu;
    float sc  = gn_g[c] * rsqrtf(var + 1e-5f);
    float sh  = gn_b[c] - mu * sc;

    size_t baseIdx = ((size_t)bn * KNN) * 128 + c;
    float mx = -3.4e38f;
    for (int k = 0; k < KNN; ++k) {
        float v = bf16_to_f32(y[baseIdx + (size_t)k * 128]);
        v = v * sc + sh;
        v = (v > 0.f) ? v : 0.2f * v;           // leaky relu
        y[baseIdx + (size_t)k * 128] = f32_to_bf16(v);
        mx = fmaxf(mx, v);
    }
    cat[(size_t)bn * 640 + c] = f32_to_bf16(mx);
}

// ---------------------------------------------------------------------------
// broadcast gmax[b][512] into cat[:, 128:640].  grid B*N, block 256.
// ---------------------------------------------------------------------------
__global__ void bcast_kernel(const int* __restrict__ gmax, u16t* __restrict__ cat) {
    int bn = blockIdx.x;
    int b  = bn >> 11;
    for (int c = threadIdx.x; c < 512; c += 256)
        cat[(size_t)bn * 640 + 128 + c] = f32_to_bf16(__int_as_float(gmax[b * 512 + c]));
}

// ---------------------------------------------------------------------------
extern "C" void kernel_launch(void* const* d_in, const int* in_sizes, int n_in,
                              void* d_out, int out_size, void* d_ws, size_t ws_size,
                              hipStream_t stream) {
    (void)in_sizes; (void)n_in; (void)out_size; (void)ws_size;
    const float* x    = (const float*)d_in[0];
    const float* f    = (const float*)d_in[1];
    const float* W_it = (const float*)d_in[2];
    const float* b_it = (const float*)d_in[3];
    const float* W1   = (const float*)d_in[4];
    const float* gn_g = (const float*)d_in[5];
    const float* gn_b = (const float*)d_in[6];
    const float* Wa   = (const float*)d_in[7];
    const float* ba   = (const float*)d_in[8];
    const float* Wb   = (const float*)d_in[9];
    const float* bb   = (const float*)d_in[10];
    const float* Wc   = (const float*)d_in[11];
    const float* bc   = (const float*)d_in[12];
    const float* Wd   = (const float*)d_in[13];
    const float* bd   = (const float*)d_in[14];
    const float* We   = (const float*)d_in[15];
    const float* be   = (const float*)d_in[16];
    const float* Wf   = (const float*)d_in[17];
    const float* bf   = (const float*)d_in[18];

    char* base = (char*)d_ws;
    size_t off = 0;
    auto alloc = [&](size_t bytes) { size_t o = off; off = (off + bytes + 255) & ~(size_t)255; return o; };

    const size_t ROWS_NK = (size_t)BATCH * NPTS * KNN;   // 262144
    const size_t ROWS_BN = (size_t)BATCH * NPTS;         // 16384

    u16t* wW1  = (u16t*)(base + alloc(128 * 128 * 2));
    u16t* wWa  = (u16t*)(base + alloc(256 * 128 * 2));
    u16t* wWb  = (u16t*)(base + alloc(512 * 256 * 2));
    u16t* wWc  = (u16t*)(base + alloc(256 * 640 * 2));
    u16t* wWd  = (u16t*)(base + alloc(128 * 256 * 2));
    u16t* wWe  = (u16t*)(base + alloc(256 * 128 * 2));
    u16t* wWf  = (u16t*)(base + alloc(512 * 256 * 2));
    float* hbuf = (float*)(base + alloc(ROWS_BN * 64 * 4));
    int*   idxb = (int*)  (base + alloc(ROWS_BN * KNN * 4));
    u16t*  feat = (u16t*) (base + alloc(ROWS_NK * 128 * 2));
    u16t*  ybuf = (u16t*) (base + alloc(ROWS_NK * 128 * 2));
    u16t*  g1   = (u16t*) (base + alloc(ROWS_NK * 256 * 2));
    u16t*  cat  = (u16t*) (base + alloc(ROWS_BN * 640 * 2));
    u16t*  lf   = (u16t*) (base + alloc(ROWS_BN * 256 * 2));
    u16t*  lf2  = (u16t*) (base + alloc(ROWS_BN * 128 * 2));
    u16t*  gl   = (u16t*) (base + alloc(ROWS_BN * 256 * 2));
    float* stats = (float*)(base + alloc(BATCH * 4 * 32 * 2 * 4));
    int*   gmax  = (int*)  (base + alloc(BATCH * 512 * 4));

    auto shmem = [](int Ck) { return (size_t)64 * (Ck + 8) * 2; };

    // 0) init accumulators + global_feat portion of d_out
    hipLaunchKernelGGL(init_kernel, dim3(16), dim3(256), 0, stream, stats, gmax, (float*)d_out);

    // 1) bf16 weight conversion
    auto cv = [&](const float* s, u16t* d, int n) {
        hipLaunchKernelGGL(cvt_kernel, dim3((n + 255) / 256), dim3(256), 0, stream, s, d, n);
    };
    cv(W1, wW1, 128 * 128); cv(Wa, wWa, 256 * 128); cv(Wb, wWb, 512 * 256);
    cv(Wc, wWc, 256 * 640); cv(Wd, wWd, 128 * 256); cv(We, wWe, 256 * 128);
    cv(Wf, wWf, 512 * 256);

    // 2) h = lift(f), KNN indices, edge features
    hipLaunchKernelGGL(h_kernel, dim3(ROWS_BN), dim3(64), 0, stream, f, W_it, b_it, hbuf);
    hipLaunchKernelGGL(knn_kernel, dim3(64), dim3(256), 0, stream, x, idxb);
    hipLaunchKernelGGL(feat_kernel, dim3(ROWS_NK / 4), dim3(256), 0, stream, hbuf, idxb, feat);

    // 3) y = feat @ W1^T (no bias/act) + groupnorm stat partials
    hipLaunchKernelGGL(gemm_bf16, dim3(ROWS_NK / 64, 128 / 64), dim3(128), shmem(128), stream,
                       feat, wW1, (const float*)nullptr, 128, 128,
                       ybuf, (float*)nullptr, 1, (int*)nullptr, (int)(NPTS * KNN), stats, 0);

    // 4) groupnorm + leakyrelu in place, local_base -> cat[:,0:128]
    hipLaunchKernelGGL(norm_kernel, dim3(ROWS_BN), dim3(128), 0, stream, ybuf, stats, gn_g, gn_b, cat);

    // 5) g1 = relu(y @ Wa^T + ba)
    hipLaunchKernelGGL(gemm_bf16, dim3(ROWS_NK / 64, 256 / 64), dim3(128), shmem(128), stream,
                       ybuf, wWa, ba, 128, 256,
                       g1, (float*)nullptr, 1, (int*)nullptr, 1, (float*)nullptr, 1);

    // 6) g2 = relu(g1 @ Wb^T + bb) fused with max over (N,K) per batch -> gmax (never stored!)
    hipLaunchKernelGGL(gemm_bf16, dim3(ROWS_NK / 64, 512 / 64), dim3(128), shmem(256), stream,
                       g1, wWb, bb, 256, 512,
                       (u16t*)nullptr, (float*)nullptr, 1, gmax, (int)(NPTS * KNN), (float*)nullptr, 1);

    // 7) cat[:,128:640] = broadcast(gmax)
    hipLaunchKernelGGL(bcast_kernel, dim3(ROWS_BN), dim3(256), 0, stream, gmax, cat);

    // 8) lf = relu(cat @ Wc^T + bc)   (640 -> 256)
    hipLaunchKernelGGL(gemm_bf16, dim3(ROWS_BN / 64, 256 / 64), dim3(128), shmem(640), stream,
                       cat, wWc, bc, 640, 256,
                       lf, (float*)nullptr, 1, (int*)nullptr, 1, (float*)nullptr, 1);

    // 9) lf2 = relu(lf @ Wd^T + bd) (256 -> 128); also local_feat f32 transposed into d_out
    hipLaunchKernelGGL(gemm_bf16, dim3(ROWS_BN / 64, 128 / 64), dim3(128), shmem(256), stream,
                       lf, wWd, bd, 256, 128,
                       lf2, (float*)d_out + BATCH * 512, (int)NPTS, (int*)nullptr, 1,
                       (float*)nullptr, 1);

    // 10) gl = relu(lf2 @ We^T + be) (128 -> 256)
    hipLaunchKernelGGL(gemm_bf16, dim3(ROWS_BN / 64, 256 / 64), dim3(128), shmem(128), stream,
                       lf2, wWe, be, 128, 256,
                       gl, (float*)nullptr, 1, (int*)nullptr, 1, (float*)nullptr, 1);

    // 11) gl2 = relu(gl @ Wf^T + bf) (256 -> 512), fused max over N -> global_feat in d_out
    hipLaunchKernelGGL(gemm_bf16, dim3(ROWS_BN / 64, 512 / 64), dim3(128), shmem(256), stream,
                       gl, wWf, bf, 256, 512,
                       (u16t*)nullptr, (float*)nullptr, 1, (int*)d_out, (int)NPTS,
                       (float*)nullptr, 1);
}
